// HyenaBlock_1400159339099
// MI455X (gfx1250) — matrix operand order repaired and checked by reading, not compiled, not run on hardware
//
#include <hip/hip_runtime.h>
#include <cstdint>

#define L_SEQ 2048
#define DIMD  1024
#define DPOSD 128
#define BATCH 4
#define NF    4096   // FFT length = 2*L

typedef __bf16 bf16_t;
typedef __attribute__((ext_vector_type(16))) __bf16 v16bf;
typedef __attribute__((ext_vector_type(8)))  __bf16 v8bf;
typedef __attribute__((ext_vector_type(8)))  float  v8f;
typedef __attribute__((ext_vector_type(4)))  unsigned int v4u;

typedef int v4i_t __attribute__((ext_vector_type(4)));
typedef __attribute__((address_space(1))) v4i_t gv4i_t;   // global int4
typedef __attribute__((address_space(3))) v4i_t lv4i_t;   // LDS int4

#if defined(__HIP_DEVICE_COMPILE__) && __has_builtin(__builtin_amdgcn_global_load_async_to_lds_b128) && __has_builtin(__builtin_amdgcn_s_wait_asynccnt)
#define HAS_ASYNC 1
#else
#define HAS_ASYNC 0
#endif

__device__ __forceinline__ void cp16_to_lds(const void* g, void* l) {
#if HAS_ASYNC
    __builtin_amdgcn_global_load_async_to_lds_b128(
        (gv4i_t*)(unsigned long long)g,
        (lv4i_t*)(unsigned int)(unsigned long long)l,
        0, 0);
#else
    *(v4u*)l = *(const v4u*)g;
#endif
}

__device__ __forceinline__ void async_wait_all() {
#if HAS_ASYNC
    __builtin_amdgcn_s_wait_asynccnt(0);
#endif
}

__device__ __forceinline__ v8f zero8() {
    v8f z;
#pragma unroll
    for (int i = 0; i < 8; ++i) z[i] = 0.0f;
    return z;
}

__device__ __forceinline__ float gelu_exact(float x) {
    return 0.5f * x * (1.0f + erff(x * 0.70710678118654752440f));
}

// ---------------------------------------------------------------------------
// Tiled bf16 WMMA GEMM:  out = epilogue( A[M,K] @ Bt[K,N] + bias[N] )
// MODE 0: fp32 store          MODE 1: gelu -> bf16 store
// MODE 2: (v)*mul[i]+add[i] -> fp32   MODE 3: (v)+add[i] -> fp32
// ---------------------------------------------------------------------------
#define BM 128
#define BN 128
#define BK 32
#define LDA_S 40    // bf16 elems per A row in LDS (80B, 16B aligned, bank-spread)
#define LDB_S 136   // bf16 elems per B row in LDS (272B)

union BF16x16 { v16bf v; v8bf h[2]; };

__device__ __forceinline__ v16bf load_fragA(const bf16_t* sa, int m0, int lane) {
    int half = lane >> 4, r = lane & 15;
    const bf16_t* row = sa + (m0 + r) * LDA_S;
    BF16x16 f;
    f.h[0] = *(const v8bf*)(row + half * 8);        // K 0..7 / 8..15
    f.h[1] = *(const v8bf*)(row + 16 + half * 8);   // K 16..23 / 24..31
    return f.v;
}

__device__ __forceinline__ v16bf load_fragB(const bf16_t* sb, int n0, int lane) {
    int half = lane >> 4, c = lane & 15;
    const bf16_t* col = sb + n0 + c;                // lane = N column, K 0..7/8..15 then 16..23/24..31
    v16bf f;
#pragma unroll
    for (int s = 0; s < 8; ++s) {
        f[s] = col[(half * 8 + s) * LDB_S];
        f[8 + s] = col[(16 + half * 8 + s) * LDB_S];
    }
    return f;
}

__device__ __forceinline__ void stage_tiles(const bf16_t* A, const bf16_t* Bt,
                                            int K, int N, int bm, int bn, int k0,
                                            bf16_t* sa, bf16_t* sb, int tid) {
#pragma unroll
    for (int c = tid; c < BM * 4; c += 256) {       // A: 128 rows x 4 x 16B
        int r = c >> 2, cc = (c & 3) * 8;
        cp16_to_lds(A + (size_t)(bm + r) * K + k0 + cc, sa + r * LDA_S + cc);
    }
#pragma unroll
    for (int c = tid; c < BK * 16; c += 256) {      // B: 32 rows x 16 x 16B
        int r = c >> 4, cc = (c & 15) * 8;
        cp16_to_lds(Bt + (size_t)(k0 + r) * N + bn + cc, sb + r * LDB_S + cc);
    }
}

template<int MODE>
__global__ __launch_bounds__(256)
void gemm_kernel(const bf16_t* __restrict__ A, const bf16_t* __restrict__ Bt,
                 const float* __restrict__ bias, void* __restrict__ out,
                 const float* __restrict__ mulp, const float* __restrict__ addp,
                 int M, int N, int K) {
    __shared__ bf16_t sA[2][BM * LDA_S];
    __shared__ bf16_t sB[2][BK * LDB_S];

    int tid = threadIdx.x;
    int bm = blockIdx.y * BM, bn = blockIdx.x * BN;
    int wave = tid >> 5, lane = tid & 31;
    int wm = (wave & 3) * 32;
    int wn = (wave >> 2) * 64;

    v8f acc[2][4];
#pragma unroll
    for (int i = 0; i < 2; ++i)
#pragma unroll
        for (int j = 0; j < 4; ++j) acc[i][j] = zero8();

    int nkt = K / BK;
    stage_tiles(A, Bt, K, N, bm, bn, 0, sA[0], sB[0], tid);
    async_wait_all();
    __syncthreads();

    int buf = 0;
    for (int kt = 0; kt < nkt; ++kt) {
        if (kt + 1 < nkt)
            stage_tiles(A, Bt, K, N, bm, bn, (kt + 1) * BK, sA[buf ^ 1], sB[buf ^ 1], tid);
        if (kt + 2 < nkt) {  // gentle L2 prefetch of the tile after next
            __builtin_prefetch(A + (size_t)(bm + (tid >> 1)) * K + (kt + 2) * BK, 0, 0);
            __builtin_prefetch(Bt + (size_t)((kt + 2) * BK + (tid & 31)) * N + bn, 0, 0);
        }
        const bf16_t* sa = sA[buf];
        const bf16_t* sb = sB[buf];
        v16bf a0 = load_fragA(sa, wm, lane);
        v16bf a1 = load_fragA(sa, wm + 16, lane);
#pragma unroll
        for (int j = 0; j < 4; ++j) {
            v16bf b = load_fragB(sb, wn + 16 * j, lane);
            acc[0][j] = __builtin_amdgcn_wmma_f32_16x16x32_bf16(
                false, a0, false, b, (short)0, acc[0][j], false, false);
            acc[1][j] = __builtin_amdgcn_wmma_f32_16x16x32_bf16(
                false, a1, false, b, (short)0, acc[1][j], false, false);
        }
        async_wait_all();
        __syncthreads();
        buf ^= 1;
    }

    // Epilogue.  C layout: lanes 0-15 => N=lane, M=j ; lanes 16-31 => N=lane-16, M=j+8
    int half = lane >> 4, nl = lane & 15;
#pragma unroll
    for (int mi = 0; mi < 2; ++mi)
#pragma unroll
        for (int ni = 0; ni < 4; ++ni) {
            int gn = bn + wn + ni * 16 + nl;
            float bv = bias[gn];
#pragma unroll
            for (int j = 0; j < 8; ++j) {
                int gm = bm + wm + mi * 16 + j + 8 * half;
                size_t idx = (size_t)gm * N + gn;
                float v = acc[mi][ni][j] + bv;
                if (MODE == 0)      ((float*)out)[idx] = v;
                else if (MODE == 1) ((bf16_t*)out)[idx] = (bf16_t)gelu_exact(v);
                else if (MODE == 2) ((float*)out)[idx] = v * mulp[idx] + addp[idx];
                else                ((float*)out)[idx] = v + addp[idx];
            }
        }
}

// ---------------------------------------------------------------------------
// LayerNorm over last dim (D=1024); one wave per row, 8 rows per block.
// ---------------------------------------------------------------------------
template<int OUT_BF>
__global__ __launch_bounds__(256)
void ln_kernel(const float* __restrict__ in, const float* __restrict__ g,
               const float* __restrict__ b, void* __restrict__ out) {
    int wave = threadIdx.x >> 5, lane = threadIdx.x & 31;
    size_t row = (size_t)blockIdx.x * 8 + wave;
    const float* x = in + row * DIMD;
    float vals[DIMD / 32];
    float s = 0.f, sq = 0.f;
#pragma unroll
    for (int i = 0; i < DIMD / 32; ++i) {
        float v = x[lane + 32 * i];
        vals[i] = v; s += v; sq += v * v;
    }
#pragma unroll
    for (int o = 16; o; o >>= 1) {
        s  += __shfl_xor(s, o, 32);
        sq += __shfl_xor(sq, o, 32);
    }
    float mu = s * (1.0f / DIMD);
    float var = sq * (1.0f / DIMD) - mu * mu;
    float rs = rsqrtf(var + 1e-5f);
#pragma unroll
    for (int i = 0; i < DIMD / 32; ++i) {
        int c = lane + 32 * i;
        float y = (vals[i] - mu) * rs * g[c] + b[c];
        if (OUT_BF) ((bf16_t*)out)[row * DIMD + c] = (bf16_t)y;
        else        ((float*)out)[row * DIMD + c] = y;
    }
}

// ---------------------------------------------------------------------------
// fp32 -> bf16 convert (plain) and transpose-convert (tiled)
// ---------------------------------------------------------------------------
__global__ __launch_bounds__(256)
void cvt_kernel(const float* __restrict__ in, bf16_t* __restrict__ out, unsigned n) {
    unsigned i = blockIdx.x * 256u + threadIdx.x;
    if (i < n) out[i] = (bf16_t)in[i];
}

__global__ __launch_bounds__(256)
void cvtT_kernel(const float* __restrict__ in, bf16_t* __restrict__ out, int R, int C) {
    __shared__ float tile[32][33];
    int bx = blockIdx.x * 32;   // C
    int by = blockIdx.y * 32;   // R
    int tx = threadIdx.x & 31, ty = threadIdx.x >> 5;
#pragma unroll
    for (int i = 0; i < 32; i += 8)
        tile[ty + i][tx] = in[(size_t)(by + ty + i) * C + bx + tx];
    __syncthreads();
#pragma unroll
    for (int i = 0; i < 32; i += 8)
        out[(size_t)(bx + ty + i) * R + by + tx] = (bf16_t)tile[tx][ty + i];
}

// window[t]*h[t,d] with transpose -> wh[d][t]
__global__ __launch_bounds__(256)
void wtr_kernel(const float* __restrict__ h, const float* __restrict__ a,
                float* __restrict__ wh) {
    __shared__ float tile[32][33];
    int bx = blockIdx.x * 32;   // d
    int by = blockIdx.y * 32;   // t
    int tx = threadIdx.x & 31, ty = threadIdx.x >> 5;
    float ea = expf(a[0]);
#pragma unroll
    for (int i = 0; i < 32; i += 8) {
        int t = by + ty + i;
        tile[ty + i][tx] = h[(size_t)t * DIMD + bx + tx] * expf(-(float)t * ea);
    }
    __syncthreads();
#pragma unroll
    for (int i = 0; i < 32; i += 8)
        wh[(size_t)(bx + ty + i) * L_SEQ + by + tx] = tile[tx][ty + i];
}

// ---------------------------------------------------------------------------
// In-LDS Stockham complex FFT, length 4096, 256 threads. sign=-1 fwd, +1 inv.
// ---------------------------------------------------------------------------
__device__ void fft_lds(float* r0, float* i0, float* r1, float* i1, int tid, float sign) {
    float *sr = r0, *si = i0, *dr = r1, *di = i1;
#pragma unroll 1
    for (int st = 0; st < 12; ++st) {
        int s = 1 << st;
        float w = sign * (6.28318530717958647692f / (float)(NF >> st));
        for (int idx = tid; idx < NF / 2; idx += 256) {
            int p = idx >> st;
            int q = idx & (s - 1);
            float sn, cs;
            __sincosf(w * (float)p, &sn, &cs);
            float ar = sr[idx],          ai = si[idx];
            float br = sr[idx + NF / 2], bi = si[idx + NF / 2];
            int o = 2 * idx - q;
            dr[o] = ar + br;  di[o] = ai + bi;
            float tr = ar - br, ti = ai - bi;
            dr[o + s] = tr * cs - ti * sn;
            di[o + s] = tr * sn + ti * cs;
        }
        __syncthreads();
        float* t;
        t = sr; sr = dr; dr = t;
        t = si; si = di; di = t;
    }
}

// Filter spectrum: WF[d] = FFT_4096( zero-padded wh[d][0..2047] )
__global__ __launch_bounds__(256)
void fft_filter_kernel(const float* __restrict__ wh,
                       float* __restrict__ WFr, float* __restrict__ WFi) {
    __shared__ float r0[NF], i0[NF], r1[NF], i1[NF];
    int d = blockIdx.x, tid = threadIdx.x;
    const float* src = wh + (size_t)d * L_SEQ;
    for (int t = tid; t < NF; t += 256) {
        r0[t] = (t < L_SEQ) ? src[t] : 0.f;
        i0[t] = 0.f;
    }
    __syncthreads();
    fft_lds(r0, i0, r1, i1, tid, -1.f);
    for (int f = tid; f < NF; f += 256) {
        WFr[(size_t)d * NF + f] = r0[f];
        WFi[(size_t)d * NF + f] = i0[f];
    }
}

// cwhz[b,:,d] = Re IFFT( FFT(zn[b,:,d] padded) * WF[d] )[0..L)
__global__ __launch_bounds__(256)
void fft_conv_kernel(const float* __restrict__ zn,
                     const float* __restrict__ WFr, const float* __restrict__ WFi,
                     float* __restrict__ cwhz) {
    __shared__ float r0[NF], i0[NF], r1[NF], i1[NF];
    int blk = blockIdx.x;
    int b = blk >> 10;          // D = 1024
    int d = blk & 1023;
    int tid = threadIdx.x;
    const float* zp = zn + ((size_t)b * L_SEQ) * DIMD + d;
    for (int t = tid; t < NF; t += 256) {
        r0[t] = (t < L_SEQ) ? zp[(size_t)t * DIMD] : 0.f;
        i0[t] = 0.f;
    }
    __syncthreads();
    fft_lds(r0, i0, r1, i1, tid, -1.f);
    const float* fr = WFr + (size_t)d * NF;
    const float* fi = WFi + (size_t)d * NF;
    for (int f = tid; f < NF; f += 256) {
        float xr = r0[f], xi = i0[f];
        float wr = fr[f], wi = fi[f];
        r0[f] = xr * wr - xi * wi;
        i0[f] = xr * wi + xi * wr;
    }
    __syncthreads();
    fft_lds(r0, i0, r1, i1, tid, 1.f);
    float* op = cwhz + ((size_t)b * L_SEQ) * DIMD + d;
    const float inv = 1.0f / (float)NF;
    for (int t = tid; t < L_SEQ; t += 256) op[(size_t)t * DIMD] = r0[t] * inv;
}

// ---------------------------------------------------------------------------
extern "C" void kernel_launch(void* const* d_in, const int* in_sizes, int n_in,
                              void* d_out, int out_size, void* d_ws, size_t ws_size,
                              hipStream_t stream) {
    (void)in_sizes; (void)n_in; (void)out_size; (void)ws_size;
    const float* z      = (const float*)d_in[0];
    const float* x      = (const float*)d_in[1];
    const float* a      = (const float*)d_in[2];
    const float* pe     = (const float*)d_in[3];
    const float* w_pos1 = (const float*)d_in[4];
    const float* b_pos1 = (const float*)d_in[5];
    const float* w_pos2 = (const float*)d_in[6];
    const float* b_pos2 = (const float*)d_in[7];
    const float* w1 = (const float*)d_in[8];
    const float* b1 = (const float*)d_in[9];
    const float* w2 = (const float*)d_in[10];
    const float* b2 = (const float*)d_in[11];
    const float* w3 = (const float*)d_in[12];
    const float* b3 = (const float*)d_in[13];
    const float* w4 = (const float*)d_in[14];
    const float* b4 = (const float*)d_in[15];
    const float* ln_g = (const float*)d_in[16];
    const float* ln_b = (const float*)d_in[17];

    char* ws = (char*)d_ws;
    // region layout (with temporal reuse):
    float* zn   = (float*)(ws + 0);                 // 33.5MB; reused as g1 later
    bf16_t* g1  = (bf16_t*)(ws + 0);
    float* WFr  = (float*)(ws + 33554432);          // 16.8MB
    float* WFi  = (float*)(ws + 50331648);          // 16.8MB
    bf16_t* g3  = (bf16_t*)(ws + 33554432);         // reuses WFr+WFi region
    float* cwhz = (float*)(ws + 67108864);          // 33.5MB; becomes x1 in-place
    bf16_t* xbf = (bf16_t*)(ws + 100663296);        // 16.8MB; reused as ln2 later
    bf16_t* ln2 = (bf16_t*)(ws + 100663296);
    float* wh   = (float*)(ws + 117440512);         // 8.4MB  [D][L]
    float* hbuf = (float*)(ws + 125829120);         // 8.4MB  [L][D]
    bf16_t* w1t = (bf16_t*)(ws + 134217728);        // [1024][2048]
    bf16_t* w2t = (bf16_t*)(ws + 138412032);        // [2048][1024]
    bf16_t* w3t = (bf16_t*)(ws + 142606336);
    bf16_t* w4t = (bf16_t*)(ws + 146800640);
    bf16_t* wp1t = (bf16_t*)(ws + 150994944);       // [128][1152]
    bf16_t* wp2t = (bf16_t*)(ws + 151289856);       // [1152][1024]
    bf16_t* pebf = (bf16_t*)(ws + 153649152);       // [2048][128]
    bf16_t* h1bf = (bf16_t*)(ws + 154173440);       // [2048][1152]

    dim3 blk(256);

    // 1) precision conversions / weight transposes
    cvt_kernel<<<dim3((BATCH * L_SEQ * DIMD) / 256), blk, 0, stream>>>(x, xbf, BATCH * L_SEQ * DIMD);
    cvt_kernel<<<dim3((L_SEQ * DPOSD) / 256), blk, 0, stream>>>(pe, pebf, L_SEQ * DPOSD);
    cvtT_kernel<<<dim3(1024 / 32, 2048 / 32), blk, 0, stream>>>(w1, w1t, 2048, 1024);
    cvtT_kernel<<<dim3(2048 / 32, 1024 / 32), blk, 0, stream>>>(w2, w2t, 1024, 2048);
    cvtT_kernel<<<dim3(1024 / 32, 2048 / 32), blk, 0, stream>>>(w3, w3t, 2048, 1024);
    cvtT_kernel<<<dim3(2048 / 32, 1024 / 32), blk, 0, stream>>>(w4, w4t, 1024, 2048);
    cvtT_kernel<<<dim3(128 / 32, 1152 / 32), blk, 0, stream>>>(w_pos1, wp1t, 1152, 128);
    cvtT_kernel<<<dim3(1152 / 32, 1024 / 32), blk, 0, stream>>>(w_pos2, wp2t, 1024, 1152);

    // 2) zn = LayerNorm(z)
    ln_kernel<0><<<dim3(BATCH * L_SEQ / 8), blk, 0, stream>>>(z, ln_g, ln_b, zn);

    // 3) positional MLP: h = (gelu(pe@wp1^T+b)) @ wp2^T + b
    gemm_kernel<1><<<dim3(1152 / BN, 2048 / BM), blk, 0, stream>>>(
        pebf, wp1t, b_pos1, h1bf, nullptr, nullptr, 2048, 1152, 128);
    gemm_kernel<0><<<dim3(1024 / BN, 2048 / BM), blk, 0, stream>>>(
        h1bf, wp2t, b_pos2, hbuf, nullptr, nullptr, 2048, 1024, 1152);

    // 4) window + transpose; filter spectra; fused FFT-conv
    wtr_kernel<<<dim3(DIMD / 32, L_SEQ / 32), blk, 0, stream>>>(hbuf, a, wh);
    fft_filter_kernel<<<dim3(DIMD), blk, 0, stream>>>(wh, WFr, WFi);
    fft_conv_kernel<<<dim3(BATCH * DIMD), blk, 0, stream>>>(zn, WFr, WFi, cwhz);

    // 5) main MLP chain with fused epilogues
    gemm_kernel<1><<<dim3(2048 / BN, 8192 / BM), blk, 0, stream>>>(
        xbf, w1t, b1, g1, nullptr, nullptr, 8192, 2048, 1024);
    gemm_kernel<2><<<dim3(1024 / BN, 8192 / BM), blk, 0, stream>>>(
        g1, w2t, b2, cwhz /*x1 in-place*/, cwhz, z, 8192, 1024, 2048);
    ln_kernel<1><<<dim3(BATCH * L_SEQ / 8), blk, 0, stream>>>(cwhz, ln_g, ln_b, ln2);
    gemm_kernel<1><<<dim3(2048 / BN, 8192 / BM), blk, 0, stream>>>(
        ln2, w3t, b3, g3, nullptr, nullptr, 8192, 2048, 1024);
    gemm_kernel<3><<<dim3(1024 / BN, 8192 / BM), blk, 0, stream>>>(
        g3, w4t, b4, d_out, nullptr, cwhz /*+x1*/, 8192, 1024, 2048);
}
